// LSTMClassifier_22239340659395
// MI455X (gfx1250) — matrix operand order repaired and checked by reading, not compiled, not run on hardware
//
#include <hip/hip_runtime.h>

// ---------------------------------------------------------------------------
// Fused LSTM (B=16384, T=512, I=2, H=48) + MLP head for MI455X (gfx1250).
// Recurrent GEMM on v_wmma_f32_16x16x32_f16; wave-private h-state in LDS
// (no intra-loop barriers); cell state + accumulators in VGPRs; input
// projection injected via the WMMA C operand; v_tanh_f32 nonlinearities with
// the sigmoid 0.5-prescale folded into the i/f/o gate weights.
// ---------------------------------------------------------------------------

typedef _Float16 h8   __attribute__((ext_vector_type(8)));
typedef _Float16 h16  __attribute__((ext_vector_type(16)));
typedef float    f8   __attribute__((ext_vector_type(8)));

#define T_LEN    512
#define HID      48
#define KPAD     64        // K padded 48 -> 64 (two 16x16x32 WMMAs)
#define NG       192       // 4*H gate columns
#define TILE_B   128       // batch rows per block (16 per wave)
#define NTHREADS 256       // 8 waves of 32

static __device__ __forceinline__ h16 cat8(h8 lo, h8 hi) {
  return __builtin_shufflevector(lo, hi, 0,1,2,3,4,5,6,7,8,9,10,11,12,13,14,15);
}

// gfx1250 native TRANS-pipe tanh (v_tanh_f32).
static __device__ __forceinline__ float fast_tanh(float x) {
#if __has_builtin(__builtin_amdgcn_tanhf)
  return __builtin_amdgcn_tanhf(x);
#elif __has_builtin(__builtin_amdgcn_tanh_f32)
  return __builtin_amdgcn_tanh_f32(x);
#else
  return 1.0f - 2.0f * __builtin_amdgcn_rcpf(1.0f + __expf(2.0f * x));
#endif
}
// sigmoid(x) where the argument is ALREADY pre-scaled by 0.5 (folded into
// the i/f/o weights): sigmoid(x) = 0.5*tanh(0.5x) + 0.5 -> fma(0.5,tanh(y),0.5)
static __device__ __forceinline__ float sig_prescaled(float y) {
  return __builtin_fmaf(0.5f, fast_tanh(y), 0.5f);
}

// One gate-column tile (N=16) over K=0..63: two chained WMMAs; the input
// projection (bias + x*W_ih) enters as the C operand of the first WMMA.
// rowp points at sWhh[(ntile*16 + lrow)*KPAD + hi*16].
static __device__ __forceinline__ f8 gate_mm(const _Float16* rowp, h16 a0, h16 a1, f8 cin) {
  h16 b0 = cat8(*(const h8*)(rowp),      *(const h8*)(rowp + 8));   // K 0..31
  h16 b1 = cat8(*(const h8*)(rowp + 32), *(const h8*)(rowp + 40));  // K 32..63
  f8 acc = __builtin_amdgcn_wmma_f32_16x16x32_f16(false, a0, false, b0, (short)0, cin, false, false);
  acc     = __builtin_amdgcn_wmma_f32_16x16x32_f16(false, a1, false, b1, (short)0, acc, false, false);
  return acc;
}

__global__ __launch_bounds__(NTHREADS) void lstm_fused_kernel(
    const float* __restrict__ seq,  const float* __restrict__ W_ih,
    const float* __restrict__ W_hh, const float* __restrict__ b_ih,
    const float* __restrict__ b_hh, const float* __restrict__ W1,
    const float* __restrict__ b1,   const float* __restrict__ W2,
    const float* __restrict__ b2,   float* __restrict__ out)
{
  // B-matrix (W_hh^T as [n][k], f16, K zero-padded to 64): 24 KB.
  // Rows of the sigmoid gates (i: 0..47, f: 48..95, o: 144..191) are
  // pre-scaled by 0.5 so sigmoid needs no leading multiply.
  __shared__ __align__(16) _Float16 sWhh[NG * KPAD];
  // h-state, f16, [row][k]; rows are wave-private: 16 KB
  __shared__ __align__(16) _Float16 sH[TILE_B * KPAD];

  const int tid = threadIdx.x;

  for (int idx = tid; idx < NG * KPAD; idx += NTHREADS) {
    int n = idx >> 6, k = idx & 63;
    float scale = (n >= 96 && n < 144) ? 1.0f : 0.5f;   // g-gate unscaled
    sWhh[idx] = (k < HID) ? (_Float16)(scale * W_hh[n * HID + k]) : (_Float16)0.0f;
  }
  for (int idx = tid; idx < TILE_B * KPAD; idx += NTHREADS)
    sH[idx] = (_Float16)0.0f;
  __syncthreads();

  const int wave  = tid >> 5;
  const int lane  = tid & 31;
  const int lrow  = lane & 15;     // N / M-row index inside a 16-wide tile
  const int hi    = lane >> 4;     // lane half (selects +8 rows / +K half)
  const int lm    = wave << 4;     // wave's local batch-row base (16 rows)
  const int gbase = blockIdx.x * TILE_B;

  // Input-projection weights for this lane: col(g,t3) = g*48 + t3*16 + lrow.
  // Same 0.5 pre-scale for the sigmoid gates (g != 2).
  float w0[4][3], w1[4][3], bsum[4][3];
  #pragma unroll
  for (int g = 0; g < 4; ++g) {
    const float scale = (g == 2) ? 1.0f : 0.5f;
    #pragma unroll
    for (int t3 = 0; t3 < 3; ++t3) {
      int col = g * HID + t3 * 16 + lrow;
      w0[g][t3]   = scale * W_ih[col * 2 + 0];
      w1[g][t3]   = scale * W_ih[col * 2 + 1];
      bsum[g][t3] = scale * (b_ih[col] + b_hh[col]);
    }
  }

  // Cell state at C-layout positions: c[t3][v] <-> row lm+v+hi*8, k t3*16+lrow
  float c[3][8];
  #pragma unroll
  for (int t3 = 0; t3 < 3; ++t3)
    #pragma unroll
    for (int v = 0; v < 8; ++v) c[t3][v] = 0.0f;

  // Per-lane sequence row offsets (float2 elements): row = gbase+lm+v+hi*8
  size_t rowoff[8];
  #pragma unroll
  for (int v = 0; v < 8; ++v)
    rowoff[v] = (size_t)(gbase + lm + v + hi * 8) * T_LEN;

  const float2* __restrict__ seq2 = (const float2*)seq;
  float2 xc[8];
  #pragma unroll
  for (int v = 0; v < 8; ++v) xc[v] = seq2[rowoff[v]];

  // A-fragment source row for this lane (wave-private h rows)
  const _Float16* hrow = &sH[(lm + lrow) * KPAD];

  for (int t = 0; t < T_LEN; ++t) {
    // Software prefetch next timestep's inputs (clamped; branchless).
    const int tn = (t + 1 < T_LEN) ? (t + 1) : t;
    float2 xn[8];
    #pragma unroll
    for (int v = 0; v < 8; ++v) xn[v] = seq2[rowoff[v] + tn];

    // A fragments for K=0..31 and K=32..63 (contiguous order, matching B:
    // lanes<16 take K [kb, kb+16); lanes>=16 take K [kb+16, kb+32)).
    h16 a0 = cat8(*(const h8*)(hrow + hi * 16),      *(const h8*)(hrow + hi * 16 + 8));
    h16 a1 = cat8(*(const h8*)(hrow + 32 + hi * 16), *(const h8*)(hrow + 40 + hi * 16));

    #pragma unroll
    for (int t3 = 0; t3 < 3; ++t3) {
      // Input projection as the WMMA C operand: xp = bias + x0*w0 + x1*w1.
      f8 xp[4];
      #pragma unroll
      for (int g = 0; g < 4; ++g) {
        #pragma unroll
        for (int v = 0; v < 8; ++v)
          xp[g][v] = __builtin_fmaf(xc[v].x, w0[g][t3],
                     __builtin_fmaf(xc[v].y, w1[g][t3], bsum[g][t3]));
      }

      // N-tile for gate g at column group t3 is n = g*3 + t3.
      f8 aI = gate_mm(&sWhh[((0 * 3 + t3) * 16 + lrow) * KPAD + hi * 16], a0, a1, xp[0]);
      f8 aF = gate_mm(&sWhh[((1 * 3 + t3) * 16 + lrow) * KPAD + hi * 16], a0, a1, xp[1]);
      f8 aG = gate_mm(&sWhh[((2 * 3 + t3) * 16 + lrow) * KPAD + hi * 16], a0, a1, xp[2]);
      f8 aO = gate_mm(&sWhh[((3 * 3 + t3) * 16 + lrow) * KPAD + hi * 16], a0, a1, xp[3]);

      // Pointwise LSTM update: fully register-local thanks to the C layout.
      #pragma unroll
      for (int v = 0; v < 8; ++v) {
        float I = sig_prescaled(aI[v]);   // weights carry the 0.5 prescale
        float F = sig_prescaled(aF[v]);
        float G = fast_tanh(aG[v]);
        float O = sig_prescaled(aO[v]);
        float cc = __builtin_fmaf(F, c[t3][v], I * G);
        c[t3][v] = cc;
        float hh = O * fast_tanh(cc);
        // Write h back for next step's A fragments (same-wave LDS is in-order).
        sH[(lm + v + hi * 8) * KPAD + t3 * 16 + lrow] = (_Float16)hh;
      }
    }

    #pragma unroll
    for (int v = 0; v < 8; ++v) xc[v] = xn[v];
  }

  __syncthreads();

  // Head: Linear(48,32) -> ReLU -> Linear(32,1); one thread per batch row.
  if (tid < TILE_B) {
    const _Float16* hr = &sH[tid * KPAD];
    float o = b2[0];
    #pragma unroll 4
    for (int j = 0; j < 32; ++j) {
      float s = b1[j];
      #pragma unroll
      for (int k = 0; k < HID; ++k) s += (float)hr[k] * W1[j * HID + k];
      o += fmaxf(s, 0.0f) * W2[j];
    }
    out[gbase + tid] = o;
  }
}

extern "C" void kernel_launch(void* const* d_in, const int* in_sizes, int n_in,
                              void* d_out, int out_size, void* d_ws, size_t ws_size,
                              hipStream_t stream) {
  const float* seq  = (const float*)d_in[0];
  const float* W_ih = (const float*)d_in[1];
  const float* W_hh = (const float*)d_in[2];
  const float* b_ih = (const float*)d_in[3];
  const float* b_hh = (const float*)d_in[4];
  const float* W1   = (const float*)d_in[5];
  const float* b1   = (const float*)d_in[6];
  const float* W2   = (const float*)d_in[7];
  const float* b2   = (const float*)d_in[8];
  float* out = (float*)d_out;

  const int B = in_sizes[0] / (T_LEN * 2);   // sequence is [B, T, 2]
  dim3 grid(B / TILE_B);
  lstm_fused_kernel<<<grid, NTHREADS, 0, stream>>>(
      seq, W_ih, W_hh, b_ih, b_hh, W1, b1, W2, b2, out);
}
